// SoftClusterAssignment_33569464386261
// MI455X (gfx1250) — matrix operand order repaired and checked by reading, not compiled, not run on hardware
//
#include <hip/hip_runtime.h>
#include <hip/hip_bf16.h>
#include <stdint.h>

// ---- problem constants (fixed by reference) ----
#define D    256      // feature dim
#define NC   65       // clusters
#define CT   5        // column tiles of 16 (padded to 80)
#define KT   8        // K tiles of 32 (256/32)
#define EPSV 1e-8f

typedef __attribute__((ext_vector_type(16))) __bf16 bf16x16;
typedef __attribute__((ext_vector_type(8)))  float  f32x8;

union Frag {
  bf16x16  v;
  uint32_t u[8];
};

// pack top-16-bits (truncated bf16) of two floats into one dword: {f1.hi16, f0.hi16}
static __device__ __forceinline__ uint32_t pack_hi_trunc(uint32_t b0, uint32_t b1) {
  // v_perm_b32: byte indices 0-3 select from S1 (=b0), 4-7 from S0 (=b1)
  return __builtin_amdgcn_perm(b1, b0, 0x07060302u);
}

// split two floats into packed truncated-bf16 hi and truncated-bf16 lo residual.
// x ~= hi + lo with |dropped| ~ 2^-16 |x|  -> fp32-quality after the 3-term GEMM.
static __device__ __forceinline__ void split_pair(float f0, float f1,
                                                  uint32_t& hi, uint32_t& lo) {
  uint32_t b0 = __float_as_uint(f0);
  uint32_t b1 = __float_as_uint(f1);
  hi = pack_hi_trunc(b0, b1);                       // 1x v_perm_b32
  float h0 = __uint_as_float(b0 & 0xFFFF0000u);     // 2x v_and
  float h1 = __uint_as_float(b1 & 0xFFFF0000u);
  uint32_t r0 = __float_as_uint(f0 - h0);           // 2x v_sub
  uint32_t r1 = __float_as_uint(f1 - h1);
  lo = pack_hi_trunc(r0, r1);                       // 1x v_perm_b32
}

// -------------------------------------------------------------------------
// Prep: pack weight into WMMA B-fragment layout (hi/lo bf16) and compute
// per-cluster reciprocal-friendly norms. One block; weight (66 KB) is
// L2-resident for the main kernel's fragment loads.
// B 32x16 layout used here (same K order as A): lane = N (col), half-waves hold K 0..15 / 16..31,
// VGPR p holds K = khalf + 2p, 2p+1  -> 16 *consecutive* K per lane.
// -------------------------------------------------------------------------
__global__ __launch_bounds__(256) void
prep_weights_kernel(const float* __restrict__ w,
                    uint32_t* __restrict__ whi,
                    uint32_t* __restrict__ wlo,
                    float* __restrict__ wn) {
  const int t = threadIdx.x;
  for (int fl = t; fl < CT * KT * 32; fl += 256) {
    const int lane = fl & 31;
    const int kt   = (fl >> 5) & 7;
    const int ct   = fl >> 8;
    const int n    = ct * 16 + (lane & 15);
    const int kb   = kt * 32 + (lane >> 4) * 16;   // 16 consecutive K

    float fv[16];
#pragma unroll
    for (int e = 0; e < 16; ++e)
      fv[e] = (n < NC) ? w[n * D + kb + e] : 0.0f;

    uint32_t hu[8], lu[8];
#pragma unroll
    for (int p = 0; p < 8; ++p)
      split_pair(fv[2 * p], fv[2 * p + 1], hu[p], lu[p]);

    const int base = fl * 8;
#pragma unroll
    for (int p = 0; p < 8; ++p) { whi[base + p] = hu[p]; wlo[base + p] = lu[p]; }
  }
  if (t < CT * 16) {
    if (t < NC) {
      float s = 0.0f;
      for (int k = 0; k < D; ++k) { float v = w[t * D + k]; s = fmaf(v, v, s); }
      wn[t] = sqrtf(s);
    } else {
      wn[t] = 1.0f;   // padded columns: harmless denom
    }
  }
}

// -------------------------------------------------------------------------
// Main: one wave per 16-row tile. Per K-tile: load 16 fp32/lane of x,
// split hi/lo to bf16 in registers, 3 bf16 WMMAs per column tile
// (hi*hi + hi*lo + lo*hi) into fp32 accumulators. Fused row norms and
// 65-wide softmax via half-wave shuffles.
// -------------------------------------------------------------------------
__global__ __launch_bounds__(256) void
soft_assign_kernel(const float* __restrict__ x,
                   const uint32_t* __restrict__ whi,
                   const uint32_t* __restrict__ wlo,
                   const float* __restrict__ wn,
                   float* __restrict__ out,
                   int nrows) {
  const int lane  = threadIdx.x & 31;
  const int wave  = threadIdx.x >> 5;
  const int tile  = blockIdx.x * 8 + wave;
  const int row0  = tile * 16;
  if (row0 >= nrows) return;

  const int rsub  = lane & 15;            // A: M = lane&15 ; C/D: N = lane&15
  const int khalf = (lane >> 4) * 16;     // A half-wave K offset

  f32x8 acc[CT];
#pragma unroll
  for (int c = 0; c < CT; ++c) acc[c] = (f32x8){0.f,0.f,0.f,0.f,0.f,0.f,0.f,0.f};

  float sq0 = 0.0f, sq1 = 0.0f;           // two chains for the row-norm FMAs
  const float* xrow = x + (size_t)(row0 + rsub) * D;

#pragma unroll
  for (int kt = 0; kt < KT; ++kt) {
    // A fragment elements: K = kt*32+khalf+e (16 consecutive K, matches B)
    float4 g0 = *(const float4*)(xrow + kt * 32 + khalf);
    float4 g1 = *(const float4*)(xrow + kt * 32 + khalf + 4);
    float4 g2 = *(const float4*)(xrow + kt * 32 + khalf + 8);
    float4 g3 = *(const float4*)(xrow + kt * 32 + khalf + 12);
    float fv[16] = {g0.x, g0.y, g0.z, g0.w, g1.x, g1.y, g1.z, g1.w,
                    g2.x, g2.y, g2.z, g2.w, g3.x, g3.y, g3.z, g3.w};

    Frag ahi, alo;
#pragma unroll
    for (int p = 0; p < 8; ++p)
      split_pair(fv[2 * p], fv[2 * p + 1], ahi.u[p], alo.u[p]);

#pragma unroll
    for (int j = 0; j < 8; ++j) {
      sq0 = fmaf(fv[2 * j],     fv[2 * j],     sq0);
      sq1 = fmaf(fv[2 * j + 1], fv[2 * j + 1], sq1);
    }

#pragma unroll
    for (int c = 0; c < CT; ++c) {
      const int fragBase = ((c * KT + kt) * 32 + lane) * 8;  // dwords
      Frag bhi, blo;
      *(uint4*)&bhi.u[0] = *(const uint4*)(whi + fragBase);
      *(uint4*)&bhi.u[4] = *(const uint4*)(whi + fragBase + 4);
      *(uint4*)&blo.u[0] = *(const uint4*)(wlo + fragBase);
      *(uint4*)&blo.u[4] = *(const uint4*)(wlo + fragBase + 4);
      acc[c] = __builtin_amdgcn_wmma_f32_16x16x32_bf16(false, ahi.v, false, bhi.v,
                                                       (short)0, acc[c], false, false);
      acc[c] = __builtin_amdgcn_wmma_f32_16x16x32_bf16(false, ahi.v, false, blo.v,
                                                       (short)0, acc[c], false, false);
      acc[c] = __builtin_amdgcn_wmma_f32_16x16x32_bf16(false, alo.v, false, bhi.v,
                                                       (short)0, acc[c], false, false);
    }
  }

  // ---- row L2 norms: halves of each row live in lane l and l^16 ----
  float sumsq = sq0 + sq1;
  sumsq += __shfl_xor(sumsq, 16, 32);
  float xn = sqrtf(sumsq);                 // lane l holds norm of row (l&15)

  float wnv[CT];
#pragma unroll
  for (int c = 0; c < CT; ++c) wnv[c] = wn[c * 16 + rsub];

  // ---- cosine sim, prob = (1+sim)^1, softmax over 65 cols ----
#pragma unroll
  for (int r = 0; r < 8; ++r) {
    const int srcLane = r + ((lane >> 4) << 3);     // row r (low half) / r+8 (high half)
    const float xnr = __shfl(xn, srcLane, 32);

    float p[CT];
    float m = -3.402823466e38f;
#pragma unroll
    for (int c = 0; c < CT; ++c) {
      float invd = __builtin_amdgcn_rcpf(fmaxf(xnr * wnv[c], EPSV)); // ~1ulp, TRANS
      float sim  = acc[c][r] * invd;
      p[c] = 1.0f + sim;                            // ALPHA=1 -> exponent 1
      if (c * 16 + rsub < NC) m = fmaxf(m, p[c]);
    }
#pragma unroll
    for (int off = 1; off < 16; off <<= 1) m = fmaxf(m, __shfl_xor(m, off, 32));

    float e[CT];
    float s = 0.0f;
#pragma unroll
    for (int c = 0; c < CT; ++c) {
      e[c] = (c * 16 + rsub < NC) ? __expf(p[c] - m) : 0.0f;
      s += e[c];
    }
#pragma unroll
    for (int off = 1; off < 16; off <<= 1) s += __shfl_xor(s, off, 32);
    const float invs = __builtin_amdgcn_rcpf(s);

    const int row = row0 + r + ((lane >> 4) << 3);
#pragma unroll
    for (int c = 0; c < CT; ++c) {
      const int col = c * 16 + rsub;
      if (col < NC) out[(size_t)row * NC + col] = e[c] * invs;
    }
  }
}

// -------------------------------------------------------------------------
extern "C" void kernel_launch(void* const* d_in, const int* in_sizes, int n_in,
                              void* d_out, int out_size, void* d_ws, size_t ws_size,
                              hipStream_t stream) {
  const float* x = (const float*)d_in[0];
  const float* w = (const float*)d_in[1];
  float* out = (float*)d_out;

  const int nrows = in_sizes[0] / D;

  uint32_t* whi = (uint32_t*)d_ws;                 // 40 KB
  uint32_t* wlo = whi + CT * KT * 32 * 8;          // 40 KB
  float*    wn  = (float*)(wlo + CT * KT * 32 * 8);// 320 B

  prep_weights_kernel<<<1, 256, 0, stream>>>(w, whi, wlo, wn);

  const int tiles  = (nrows + 15) / 16;
  const int blocks = (tiles + 7) / 8;
  soft_assign_kernel<<<blocks, 256, 0, stream>>>(x, whi, wlo, wn, out, nrows);
}
